// FrameInterpolate_14877766714082
// MI455X (gfx1250) — matrix operand order repaired and checked by reading, not compiled, not run on hardware
//
#include <hip/hip_runtime.h>
#include <stdint.h>

typedef __attribute__((ext_vector_type(16))) _Float16 v16h;
typedef __attribute__((ext_vector_type(2)))  _Float16 v2h;
typedef __attribute__((ext_vector_type(8)))  float    v8f;
typedef __attribute__((ext_vector_type(2)))  float    v2f;

#define B_ 8
#define H_ 512
#define W_ 512

// ---------------------------------------------------------------------------
// Kernel 1: bilinear warps + concat -> feats[b,y,x,10] as f16 (5 x half2)
// feats channels: [prv_w(3), nxt_w(3), flo01(2), flo10(2)]
// ---------------------------------------------------------------------------
__global__ __launch_bounds__(256) void feats_kernel(
    const float* __restrict__ prv, const float* __restrict__ nxt,
    const float* __restrict__ flo01, const float* __restrict__ flo10,
    _Float16* __restrict__ feats) {
  int idx = blockIdx.x * blockDim.x + threadIdx.x;
  const int total = B_ * H_ * W_;
  if (idx >= total) return;
  int x = idx % W_;
  int y = (idx / W_) % H_;
  int b = idx / (W_ * H_);

  float f01x = flo01[(size_t)idx * 2 + 0];
  float f01y = flo01[(size_t)idx * 2 + 1];
  float f10x = flo10[(size_t)idx * 2 + 0];
  float f10y = flo10[(size_t)idx * 2 + 1];

  float o[10];

  // bilinear sample with tfa clamping semantics
  auto sample = [&](const float* __restrict__ img, float qx, float qy,
                    float* out3) {
    float y0 = floorf(qy);
    float x0 = floorf(qx);
    y0 = fminf(fmaxf(y0, 0.0f), (float)(H_ - 2));
    x0 = fminf(fmaxf(x0, 0.0f), (float)(W_ - 2));
    float ay = fminf(fmaxf(qy - y0, 0.0f), 1.0f);
    float ax = fminf(fmaxf(qx - x0, 0.0f), 1.0f);
    int yi = (int)y0, xi = (int)x0;
    const float* p00 = img + (((size_t)b * H_ + yi) * W_ + xi) * 3;
    const float* p01 = p00 + 3;
    const float* p10 = p00 + (size_t)W_ * 3;
    const float* p11 = p10 + 3;
#pragma unroll
    for (int c = 0; c < 3; ++c) {
      float top = p00[c] * (1.0f - ax) + p01[c] * ax;
      float bot = p10[c] * (1.0f - ax) + p11[c] * ax;
      out3[c] = top * (1.0f - ay) + bot * ay;
    }
  };

  // prv warped by 0.5*flo10 ; nxt warped by 0.5*flo01
  sample(prv, (float)x + 0.5f * f10x, (float)y + 0.5f * f10y, &o[0]);
  sample(nxt, (float)x + 0.5f * f01x, (float)y + 0.5f * f01y, &o[3]);
  o[6] = f01x; o[7] = f01y;
  o[8] = f10x; o[9] = f10y;

  // packed half2 stores (20B per pixel, 4B aligned)
  v2h* dst = (v2h*)(feats + (size_t)idx * 10);
#pragma unroll
  for (int c = 0; c < 5; ++c) {
    v2h p;
    p[0] = (_Float16)o[2 * c];
    p[1] = (_Float16)o[2 * c + 1];
    dst[c] = p;
  }
}

// fast mish on a float2: x * tanh(softplus(x)) = x * e(e+2)/(e(e+2)+2), e=e^x.
// Packed v_pk_* for the ALU part; scalar v_exp_f32/v_rcp_f32 for the trans part.
__device__ __forceinline__ v2f fast_mish2(v2f v) {
  const float LOG2E = 1.4426950408889634f;
  v2f arg = v * LOG2E;                                    // pk_mul
  arg = __builtin_elementwise_min(arg, v2f{57.7f, 57.7f});  // pk_min (clamp e^40)
  v2f e;
  e.x = __builtin_amdgcn_exp2f(arg.x);                    // v_exp_f32
  e.y = __builtin_amdgcn_exp2f(arg.y);
  v2f num = e * (e + 2.0f);                               // pk_add + pk_mul
  v2f den = num + 2.0f;                                   // pk_add
  v2f r;
  r.x = __builtin_amdgcn_rcpf(den.x);                     // v_rcp_f32
  r.y = __builtin_amdgcn_rcpf(den.y);
  return v * num * r;                                     // 2x pk_mul
}

// ---------------------------------------------------------------------------
// Kernel 2: per 16x16-pixel tile: depthwise 3x3 (f32) -> WMMA 10->64
// (bias in C, Mish) -> transposed WMMA 64->3 (bias in C) -> coalesced stores
// ---------------------------------------------------------------------------
__global__ __launch_bounds__(256) void fused_conv_kernel(
    const _Float16* __restrict__ feats, const float* __restrict__ dwk,
    const float* __restrict__ pwk, const float* __restrict__ b1,
    const float* __restrict__ w2, const float* __restrict__ b2,
    float* __restrict__ out) {
  __shared__ uint32_t sFeatU[18 * 18 * 5];  // halo tile, 5 dwords / entry
  __shared__ _Float16 sDw[256 * 32];        // depthwise out, K padded to 32
  __shared__ _Float16 sY[256 * 64];         // mish output [pixel][k]
  __shared__ v16h sPwFrag[4][32];           // mm1 B fragments [nt][lane]
  __shared__ v16h sW2FragT[2][32];          // mm2 A fragments (W2^T) [kc][lane]
  __shared__ float sB1[64];
  __shared__ float sDwK[9 * 10];
  __shared__ float sB2[4];

  const int tid = threadIdx.x;
  const int tilesX = W_ / 16;
  const int tilesY = H_ / 16;
  const int blk = blockIdx.x;
  const int tx0 = (blk % tilesX) * 16;
  const int ty0 = ((blk / tilesX) % tilesY) * 16;
  const int b = blk / (tilesX * tilesY);

  // --- stage weights into LDS, pre-swizzled into WMMA fragment layouts ---
  if (tid < 4 * 32) {
    // matmul1 B-frag: lane l, elem j -> B[k][n], k=(l<16? j:16+j), n=l&15
    int nt = tid >> 5, l = tid & 31;
    int n = nt * 16 + (l & 15);
    v16h t;
#pragma unroll
    for (int j = 0; j < 16; ++j) {
      int k = (l < 16) ? j : (16 + j);
      t[j] = (k < 10) ? (_Float16)pwk[k * 64 + n] : (_Float16)0.0f;
    }
    sPwFrag[nt][l] = t;
  } else if (tid < 4 * 32 + 2 * 32) {
    // matmul2 A-frag of W2^T (16 rows = out-channels, K=64 over 2 chunks):
    // lane l, elem j -> A[n][k], n=l&15,
    // k = kc*32 + 2*kG + j, kG = (l>=16)?8:0
    int i = tid - 4 * 32;
    int kc = i >> 5, l = i & 31;
    int n = l & 15;
    int kG = (l >= 16) ? 8 : 0;
    v16h t;
#pragma unroll
    for (int j = 0; j < 16; ++j) {
      int k = kc * 32 + 2 * kG + j;
      t[j] = (n < 3) ? (_Float16)w2[k * 3 + n] : (_Float16)0.0f;
    }
    sW2FragT[kc][l] = t;
  } else if (tid < 192 + 64) {
    sB1[tid - 192] = b1[tid - 192];
  }
  if (tid < 90) sDwK[tid] = dwk[tid];  // (kh*3+kw)*10 + c
  if (tid >= 96 && tid < 100) sB2[tid - 96] = (tid < 99) ? b2[tid - 96] : 0.0f;

  // --- load 18x18x10 feats halo as dwords (zero-pad outside image) ---
  for (int i = tid; i < 18 * 18; i += 256) {
    int hy = i / 18, hx = i % 18;
    int gy = ty0 + hy - 1, gx = tx0 + hx - 1;
    bool inb = (gy >= 0) && (gy < H_) && (gx >= 0) && (gx < W_);
    const uint32_t* src =
        (const uint32_t*)(feats + (((size_t)b * H_ + gy) * W_ + gx) * 10);
    uint32_t* dst = sFeatU + i * 5;
#pragma unroll
    for (int c = 0; c < 5; ++c) dst[c] = inb ? src[c] : 0u;
  }
  __syncthreads();

  // --- depthwise 3x3 in f32, one pixel per thread -> sDw[pixel][0..31] ---
  {
    const _Float16* sFeatH = (const _Float16*)sFeatU;
    int px = tid & 15, py = tid >> 4;
    _Float16* dst = sDw + tid * 32;
#pragma unroll
    for (int c = 0; c < 10; ++c) {
      float acc = 0.0f;
#pragma unroll
      for (int kh = 0; kh < 3; ++kh)
#pragma unroll
        for (int kw = 0; kw < 3; ++kw)
          acc += (float)sFeatH[((py + kh) * 18 + (px + kw)) * 10 + c] *
                 sDwK[(kh * 3 + kw) * 10 + c];
      dst[c] = (_Float16)acc;
    }
#pragma unroll
    for (int c = 10; c < 32; ++c) dst[c] = (_Float16)0.0f;
  }
  __syncthreads();

  const int wave = tid >> 5;
  const int lane = tid & 31;
  const int laneN = lane & 15;
  const int kGroup = (lane >= 16) ? 8 : 0;  // A-frag K half select
  const int mHi = (lane >= 16) ? 8 : 0;     // C/D row-half select

  // --- matmul1: [256 x 32] x [32 x 64]; bias in C; Mish -> sY[pixel][k] ---
  for (int mi = 0; mi < 2; ++mi) {
    const int mt = wave * 2 + mi;
    const int m = mt * 16 + laneN;
    v16h a;
    const _Float16* arow = sDw + m * 32;
#pragma unroll
    for (int j = 0; j < 16; ++j) {  // contiguous: 2x ds_load_b128
      int k = 2 * kGroup + j;
      a[j] = arow[k];
    }
#pragma unroll
    for (int nt = 0; nt < 4; ++nt) {
      v16h bf = sPwFrag[nt][lane];  // contiguous 32B
      const int nn = nt * 16 + laneN;
      const float bias = sB1[nn];
      v8f c;
#pragma unroll
      for (int r = 0; r < 8; ++r) c[r] = bias;  // D = A*B + bias
      c = __builtin_amdgcn_wmma_f32_16x16x32_f16(false, a, false, bf,
                                                 (short)0, c, false, false);
#pragma unroll
      for (int r2 = 0; r2 < 4; ++r2) {
        v2f pv = {c[2 * r2], c[2 * r2 + 1]};
        v2f mv = fast_mish2(pv);
        int mm = mt * 16 + 2 * r2 + mHi;
        sY[mm * 64 + nn] = (_Float16)mv.x;
        sY[(mm + 1) * 64 + nn] = (_Float16)mv.y;
      }
    }
  }
  __syncthreads();

  // --- matmul2 (transposed): out^T[n][m] = W2^T[n][k] * Y^T[k][m] + b2 ---
  // B-frag: lane l, elem j -> Y^T[k][m] = sY[m][k], m = mt*16 + (l&15),
  //         k = kc*32 + (l<16 ? j : 16+j)  -> one contiguous 32B LDS read.
  for (int mi = 0; mi < 2; ++mi) {
    const int mt = wave * 2 + mi;
    const int m = mt * 16 + laneN;
    v8f c;
#pragma unroll
    for (int r = 0; r < 8; ++r)  // rows = out channels; bias in C
      c[r] = ((r + mHi) < 3) ? sB2[r] : 0.0f;
#pragma unroll
    for (int kc = 0; kc < 2; ++kc) {
      const v16h* bp =
          (const v16h*)(sY + m * 64 + kc * 32 + ((lane < 16) ? 0 : 16));
      v16h bf = *bp;                       // 2x ds_load_b128
      v16h af = sW2FragT[kc][lane];        // contiguous 32B
      c = __builtin_amdgcn_wmma_f32_16x16x32_f16(false, af, false, bf,
                                                 (short)0, c, false, false);
    }
    // lane<16 owns channels 0..7 of pixel m; write channels 0..2 coalesced
    if (lane < 16) {
      int py = m >> 4, px = m & 15;
      float* o = out + (((size_t)b * H_ + (ty0 + py)) * W_ + (tx0 + px)) * 3;
      o[0] = c[0];
      o[1] = c[1];
      o[2] = c[2];
    }
  }
}

// ---------------------------------------------------------------------------
extern "C" void kernel_launch(void* const* d_in, const int* in_sizes, int n_in,
                              void* d_out, int out_size, void* d_ws,
                              size_t ws_size, hipStream_t stream) {
  (void)in_sizes; (void)n_in; (void)out_size; (void)ws_size;
  const float* prv   = (const float*)d_in[0];
  const float* nxt   = (const float*)d_in[1];
  const float* flo01 = (const float*)d_in[2];
  const float* flo10 = (const float*)d_in[3];
  const float* dwk   = (const float*)d_in[4];
  const float* pwk   = (const float*)d_in[5];
  const float* b1    = (const float*)d_in[6];
  const float* w2    = (const float*)d_in[7];
  const float* b2    = (const float*)d_in[8];
  float* out = (float*)d_out;

  _Float16* feats = (_Float16*)d_ws;  // B*H*W*10 f16 = ~42 MB

  const int total = B_ * H_ * W_;
  feats_kernel<<<(total + 255) / 256, 256, 0, stream>>>(prv, nxt, flo01, flo10,
                                                        feats);

  const int nBlocks = B_ * (H_ / 16) * (W_ / 16);  // 8192
  fused_conv_kernel<<<nBlocks, 256, 0, stream>>>(feats, dwk, pwk, b1, w2, b2,
                                                 out);
}
